// DiscreteDiT_52269751992405
// MI455X (gfx1250) — matrix operand order repaired and checked by reading, not compiled, not run on hardware
//
#include <hip/hip_runtime.h>
#include <math.h>

// ---------------- model dims ----------------
#define BSZ 8
#define NN  192
#define DD  256
#define ENCD 1728
#define HD  64
#define NROWS (BSZ*NN)        // 1536
#define NNSQ (NN*NN)          // 36864

typedef __attribute__((ext_vector_type(16))) _Float16 v16h;
typedef __attribute__((ext_vector_type(8)))  _Float16 v8h;
typedef __attribute__((ext_vector_type(4)))  _Float16 v4h;
typedef __attribute__((ext_vector_type(8)))  float    v8f;

// CDNA5 async global->LDS copy path (guarded: falls back to plain loads if the
// toolchain doesn't expose the builtins; host pass always takes the fallback).
#if defined(__HIP_DEVICE_COMPILE__) && \
    __has_builtin(__builtin_amdgcn_global_load_async_to_lds_b128) && \
    __has_builtin(__builtin_amdgcn_s_wait_asynccnt)
#define HAVE_ASYNC_COPY 1
#define AS1 __attribute__((address_space(1)))
#define AS3 __attribute__((address_space(3)))
typedef int v4i_a __attribute__((vector_size(16)));   // matches builtin param type
#endif

static __device__ __forceinline__ float geluf(float x){
  float x3 = x*x*x;
  return 0.5f*x*(1.0f + tanhf(0.7978845608028654f*(x + 0.044715f*x3)));
}
static __device__ __forceinline__ float siluf(float x){
  return x / (1.0f + expf(-x));
}

static __device__ __forceinline__ v8f wmma16(v16h a, v16h b, v8f c){
  return __builtin_amdgcn_wmma_f32_16x16x32_f16(false, a, false, b, (short)0, c, false, false);
}

// A fragment: 16x32 (MxK), f16 row-major source (ld in halfs).
// lanes 0-15 row M=lane, K = e + 8*(e>=8); lanes 16-31 K shifted +8
// => two contiguous b128 LDS loads per lane.
static __device__ __forceinline__ v16h load_a_frag(const _Float16* base, int ld, int m0, int k0, int lane){
  const _Float16* p = base + (size_t)(m0 + (lane & 15)) * ld + k0 + ((lane >> 4) << 3);
  v8h lo = *(const v8h*)(p);
  v8h hi = *(const v8h*)(p + 16);
  v16h a;
#pragma unroll
  for (int e = 0; e < 8; ++e){ a[e] = lo[e]; a[e+8] = hi[e]; }
  return a;
}

// B fragment: 32x16 (KxN), from a K-major (transposed) f16 buffer (ld in halfs).
// N = lane&15; K = e + 8*(e>=8) + 8*(lane>=16) => two contiguous b128 loads per lane.
static __device__ __forceinline__ v16h load_b_fragT(const _Float16* baseT, int ld, int n0, int k0, int lane){
  const _Float16* p = baseT + (size_t)(n0 + (lane & 15)) * ld + k0 + ((lane >> 4) << 3);
  v8h lo = *(const v8h*)(p);
  v8h hi = *(const v8h*)(p + 16);
  v16h b;
#pragma unroll
  for (int e = 0; e < 8; ++e){ b[e] = lo[e]; b[e+8] = hi[e]; }
  return b;
}

static __device__ __forceinline__ float block_reduce256(float v, float* red, int tid, int ismax){
  red[tid] = v;
  __syncthreads();
  for (int s = 128; s > 0; s >>= 1){
    if (tid < s) red[tid] = ismax ? fmaxf(red[tid], red[tid+s]) : (red[tid] + red[tid+s]);
    __syncthreads();
  }
  float r = red[0];
  __syncthreads();
  return r;
}

// ------------- generic WMMA GEMM: C[M,N] = epilogue(A[M,K] @ W[K,N] + bias) -------------
// mode 0: +bias        mode 1: +bias +pos_emb[row%192, col]
// mode 2: resid[row,col] + gate[b*1536+col]*( +bias )    mode 3: gelu(+bias)
static __device__ __forceinline__ void store_tile(v8f acc, int row0, int col0, int lane,
    float* C, int Nld, const float* bias, int mode, const float* aux1, const float* aux2)
{
  int col = col0 + (lane & 15);
  int rbase = row0 + ((lane >> 4) << 3);
  float bv = bias ? bias[col] : 0.0f;
#pragma unroll
  for (int r = 0; r < 8; ++r){
    int row = rbase + r;
    size_t o = (size_t)row * Nld + col;
    float x = acc[r] + bv;
    if (mode == 1)      x += aux1[(size_t)(row % NN) * Nld + col];
    else if (mode == 2) x = aux2[o] + aux1[(row / NN) * (6*DD) + col] * x;
    else if (mode == 3) x = geluf(x);
    C[o] = x;
  }
}

__global__ void k_gemm(const float* __restrict__ A, const float* __restrict__ W,
                       const float* __restrict__ bias, float* C,
                       int M, int K, int N, int mode,
                       const float* __restrict__ aux1, const float* aux2)
{
  // fp32 emulated via f16 hi/lo split: AB ~= AhBh + AhBl + AlBh
  __shared__ _Float16 Ah[2048], Al[2048], Bh[2048], Bl[2048];
  const int tid = threadIdx.x;
  const int lane = tid & 31, wave = tid >> 5;
  const int m0 = blockIdx.y * 64, n0 = blockIdx.x * 64;
  const int t0 = wave * 2, t1 = t0 + 1;
  const int mt0 = (t0 >> 2) << 4, nt0 = (t0 & 3) << 4;
  const int mt1 = (t1 >> 2) << 4, nt1 = (t1 & 3) << 4;
  v8f acc0 = {}, acc1 = {};
  for (int kb = 0; kb < K; kb += 32){
    // stage A block 64x32 as float4 loads -> packed half4 hi/lo LDS stores
#pragma unroll
    for (int i = 0; i < 2; ++i){
      int f4 = tid + (i << 8);                 // 0..511
      int ml = f4 >> 3, k4 = (f4 & 7) << 2;    // coalesced along K
      const float4 av = *(const float4*)&A[(size_t)(m0 + ml) * K + kb + k4];
      float fx[4] = {av.x, av.y, av.z, av.w};
      v4h hi, lo;
#pragma unroll
      for (int q = 0; q < 4; ++q){
        _Float16 hv = (_Float16)fx[q];
        hi[q] = hv; lo[q] = (_Float16)(fx[q] - (float)hv);
      }
      *(v4h*)&Ah[ml*32 + k4] = hi;
      *(v4h*)&Al[ml*32 + k4] = lo;
    }
    // stage B block 32x64 transposed: 4 coalesced b32 loads -> packed b64 stores
#pragma unroll
    for (int i = 0; i < 2; ++i){
      int f4 = tid + (i << 8);                 // 0..511
      int nl = f4 & 63, k4 = (f4 >> 6) << 2;   // coalesced along N
      size_t gbase = (size_t)(kb + k4) * N + n0 + nl;
      v4h hi, lo;
#pragma unroll
      for (int q = 0; q < 4; ++q){
        float wv = W[gbase + (size_t)q * N];
        _Float16 hv = (_Float16)wv;
        hi[q] = hv; lo[q] = (_Float16)(wv - (float)hv);
      }
      *(v4h*)&Bh[nl*32 + k4] = hi;
      *(v4h*)&Bl[nl*32 + k4] = lo;
    }
    if (kb + 32 < K) __builtin_prefetch(&A[(size_t)(m0 + (tid >> 2)) * K + kb + 32], 0, 1);
    __syncthreads();
    {
      v16h ah = load_a_frag(Ah, 32, mt0, 0, lane);
      v16h al = load_a_frag(Al, 32, mt0, 0, lane);
      v16h bh = load_b_fragT(Bh, 32, nt0, 0, lane);
      v16h bl = load_b_fragT(Bl, 32, nt0, 0, lane);
      acc0 = wmma16(ah, bh, acc0);
      acc0 = wmma16(ah, bl, acc0);
      acc0 = wmma16(al, bh, acc0);
      ah = load_a_frag(Ah, 32, mt1, 0, lane);
      al = load_a_frag(Al, 32, mt1, 0, lane);
      bh = load_b_fragT(Bh, 32, nt1, 0, lane);
      bl = load_b_fragT(Bl, 32, nt1, 0, lane);
      acc1 = wmma16(ah, bh, acc1);
      acc1 = wmma16(ah, bl, acc1);
      acc1 = wmma16(al, bh, acc1);
    }
    __syncthreads();
  }
  store_tile(acc0, m0 + mt0, n0 + nt0, lane, C, N, bias, mode, aux1, aux2);
  store_tile(acc1, m0 + mt1, n0 + nt1, lane, C, N, bias, mode, aux1, aux2);
}

// ------------- timestep embedding + t-MLP (fp32, tiny) -------------
__global__ void k_timestep(const int* __restrict__ ts,
                           const float* __restrict__ w1, const float* __restrict__ b1,
                           const float* __restrict__ w2, const float* __restrict__ b2,
                           float* __restrict__ tout, float* __restrict__ st)
{
  __shared__ float emb[DD];
  __shared__ float t1s[DD];
  int b = blockIdx.x, c = threadIdx.x;
  float tv = (float)ts[b];
  if (c < 128){
    float f = expf(-9.210340371976184f * (float)c / 128.0f);
    float a = tv * f;
    emb[c] = sinf(a);
    emb[c + 128] = cosf(a);
  }
  __syncthreads();
  float acc = b1[c];
  for (int k2 = 0; k2 < DD; ++k2) acc += emb[k2] * w1[k2*DD + c];
  t1s[c] = siluf(acc);
  __syncthreads();
  float acc2 = b2[c];
  for (int k2 = 0; k2 < DD; ++k2) acc2 += t1s[k2] * w2[k2*DD + c];
  tout[b*DD + c] = acc2;
  st[b*DD + c]  = siluf(acc2);
}

// out[b,j] = st[b,:] @ W[:,j] + bias[j]   (B=8 rows, K=256)
__global__ void k_smallmm(const float* __restrict__ S, const float* __restrict__ W,
                          const float* __restrict__ bias, float* __restrict__ out, int Ncols)
{
  int idx = blockIdx.x * blockDim.x + threadIdx.x;
  if (idx >= BSZ * Ncols) return;
  int b = idx / Ncols, j = idx - b * Ncols;
  float acc = bias[j];
  for (int k2 = 0; k2 < DD; ++k2) acc += S[b*DD + k2] * W[k2*Ncols + j];
  out[idx] = acc;
}

// hn = LN(h_row) * (1 + scale[b,c]) + shift[b,c]
__global__ void k_ln_mod(const float* __restrict__ h, float* __restrict__ hn,
                         const float* __restrict__ shift, const float* __restrict__ scale,
                         int modstride)
{
  __shared__ float red[256];
  int row = blockIdx.x, c = threadIdx.x;
  int b = row / NN;
  float x = h[(size_t)row * DD + c];
  float s1 = block_reduce256(x,     red, c, 0);
  float s2 = block_reduce256(x * x, red, c, 0);
  float mean = s1 * (1.0f/256.0f);
  float var  = s2 * (1.0f/256.0f) - mean * mean;
  float y = (x - mean) * rsqrtf(var + 1e-6f);
  hn[(size_t)row*DD + c] = y * (1.0f + scale[b*modstride + c]) + shift[b*modstride + c];
}

// one block per (b,head); 192 threads, each owns one query row (online softmax, fp32)
__global__ void k_attn(const float* __restrict__ Q, const float* __restrict__ Kk,
                       const float* __restrict__ Vv, float* __restrict__ O)
{
  extern __shared__ char smem_attn[];
  float* ks = (float*)smem_attn;           // [192*64]
  float* vs = ks + NN*HD;                  // [192*64]
  int bh = blockIdx.x;
  int b = bh >> 2, hidx = bh & 3;
  int i = threadIdx.x;
  size_t base = ((size_t)b * NN) * DD + (size_t)hidx * HD;
#if HAVE_ASYNC_COPY
  // pure byte copy -> CDNA5 async VMEM->LDS path (tracked on ASYNCcnt)
  for (int f4 = i; f4 < (NN*HD)/4; f4 += NN){
    int j = f4 >> 4, c4 = (f4 & 15) << 2;
    __builtin_amdgcn_global_load_async_to_lds_b128(
        (AS1 v4i_a*)(Kk + base + (size_t)j*DD + c4),
        (AS3 v4i_a*)(ks + (j<<6) + c4), 0, 0);
    __builtin_amdgcn_global_load_async_to_lds_b128(
        (AS1 v4i_a*)(Vv + base + (size_t)j*DD + c4),
        (AS3 v4i_a*)(vs + (j<<6) + c4), 0, 0);
  }
  __builtin_amdgcn_s_wait_asynccnt(0);
#else
  for (int f4 = i; f4 < (NN*HD)/4; f4 += NN){
    int j = f4 >> 4, c4 = (f4 & 15) << 2;
    *(float4*)&ks[(j<<6) + c4] = *(const float4*)&Kk[base + (size_t)j*DD + c4];
    *(float4*)&vs[(j<<6) + c4] = *(const float4*)&Vv[base + (size_t)j*DD + c4];
  }
#endif
  __syncthreads();
  float qr[HD];
#pragma unroll
  for (int c = 0; c < HD; ++c) qr[c] = Q[base + (size_t)i*DD + c];
  float m = -1e30f, s = 0.0f;
  float oa[HD];
#pragma unroll
  for (int c = 0; c < HD; ++c) oa[c] = 0.0f;
  for (int j = 0; j < NN; ++j){
    float d = 0.0f;
#pragma unroll
    for (int c = 0; c < HD; ++c) d += qr[c] * ks[j*HD + c];
    d *= 0.125f;                            // 1/sqrt(64)
    float mn = fmaxf(m, d);
    float corr = expf(m - mn);
    float p = expf(d - mn);
    s = s * corr + p;
#pragma unroll
    for (int c = 0; c < HD; ++c) oa[c] = oa[c] * corr + p * vs[j*HD + c];
    m = mn;
  }
  float inv = 1.0f / s;
#pragma unroll
  for (int c = 0; c < HD; ++c) O[base + (size_t)i*DD + c] = oa[c] * inv;
}

// pre-pack pair layer-2 weights [256,256] fp32 -> K-major f16 hi/lo
__global__ void k_pack_w2(const float* __restrict__ W, _Float16* __restrict__ WTh,
                          _Float16* __restrict__ WTl)
{
  int idx = blockIdx.x * 256 + threadIdx.x;   // 65536 elems
  int n = idx >> 8, k2 = idx & 255;
  float x = W[k2*256 + n];
  _Float16 hv = (_Float16)x;
  WTh[n*256 + k2] = hv;
  WTl[n*256 + k2] = (_Float16)(x - (float)hv);
}

// one block per (b,i): h1[j,:]=gelu(u_i+v_j) in LDS (f16 hi/lo), WMMA vs W2,
// gelu(+b2), dot with w_out via ds_add_f32 -> Lraw[b,i,:]
__global__ void k_pair(const float* __restrict__ U, const float* __restrict__ V,
                       const _Float16* __restrict__ W2h, const _Float16* __restrict__ W2l,
                       const float* __restrict__ b2, const float* __restrict__ wout,
                       const float* __restrict__ bout, float* __restrict__ Lraw)
{
  extern __shared__ char smem_pair[];
  _Float16* h1h = (_Float16*)smem_pair;       // [96*256]
  _Float16* h1l = h1h + 96*256;               // [96*256]
  float* lred   = (float*)(h1l + 96*256);     // [192]
  const int tid = threadIdx.x, lane = tid & 31, wave = tid >> 5;
  const int b = blockIdx.x / NN, i = blockIdx.x % NN;
  const float* urow  = U + ((size_t)b*NN + i) * DD;
  const float* vbase = V + (size_t)b * NN * DD;
  if (tid < NN) lred[tid] = 0.0f;
  for (int jb = 0; jb < NN; jb += 96){
    __syncthreads();
    for (int f4 = tid; f4 < 96*64; f4 += 256){
      int j = f4 >> 6, c4 = (f4 & 63) << 2;
      const float4 u4 = *(const float4*)&urow[c4];
      const float4 w4 = *(const float4*)&vbase[(size_t)(jb + j)*DD + c4];
      float fx[4] = {u4.x + w4.x, u4.y + w4.y, u4.z + w4.z, u4.w + w4.w};
      v4h hi, lo;
#pragma unroll
      for (int q = 0; q < 4; ++q){
        float x = geluf(fx[q]);
        _Float16 hv = (_Float16)x;
        hi[q] = hv; lo[q] = (_Float16)(x - (float)hv);
      }
      *(v4h*)&h1h[(j<<8) + c4] = hi;
      *(v4h*)&h1l[(j<<8) + c4] = lo;
    }
    __syncthreads();
    for (int s = 0; s < 12; ++s){
      int tile = wave + 8*s;                  // 0..95  (6 mt x 16 nt)
      int mt = (tile >> 4) << 4;
      int nt = (tile & 15) << 4;
      v8f acc = {};
      for (int kb2 = 0; kb2 < 256; kb2 += 32){
        v16h ah = load_a_frag(h1h, 256, mt, kb2, lane);
        v16h al = load_a_frag(h1l, 256, mt, kb2, lane);
        v16h bh = load_b_fragT(W2h, 256, nt, kb2, lane);
        v16h bl = load_b_fragT(W2l, 256, nt, kb2, lane);
        acc = wmma16(ah, bh, acc);
        acc = wmma16(ah, bl, acc);
        acc = wmma16(al, bh, acc);
      }
      int col = nt + (lane & 15);
      float bb = b2[col], wo = wout[col];
      int rbase = jb + mt + ((lane >> 4) << 3);
#pragma unroll
      for (int r = 0; r < 8; ++r){
        float h2 = geluf(acc[r] + bb);
        atomicAdd(&lred[rbase + r], h2 * wo);
      }
    }
  }
  __syncthreads();
  if (tid < NN)
    Lraw[((size_t)b*NN + i) * NN + tid] = lred[tid] + bout[0];
}

// symmetrize + diag mask + write logits output + la = logits / 0.05
__global__ void k_sym(const float* __restrict__ Lraw, float* __restrict__ logits,
                      float* __restrict__ la)
{
  int idx = blockIdx.x * 256 + threadIdx.x;
  if (idx >= BSZ * NNSQ) return;
  int b = idx / NNSQ, r = idx - b * NNSQ;
  int i2 = r / NN, j = r - i2 * NN;
  float v = 0.5f * (Lraw[idx] + Lraw[(size_t)b*NNSQ + (size_t)j*NN + i2]);
  if (i2 == j) v = -1e9f;
  logits[idx] = v;
  la[idx] = v / 0.05f;
}

__global__ void k_sink_row(float* __restrict__ la){
  __shared__ float red[256];
  int row = blockIdx.x, tid = threadIdx.x;
  float v = (tid < NN) ? la[(size_t)row*NN + tid] : -1e30f;
  float mx = block_reduce256(v, red, tid, 1);
  float e = (tid < NN) ? expf(v - mx) : 0.0f;
  float s = block_reduce256(e, red, tid, 0);
  float lse = mx + logf(s);
  if (tid < NN) la[(size_t)row*NN + tid] = v - lse;
}

__global__ void k_sink_col(float* __restrict__ la){
  __shared__ float red[256];
  int bj = blockIdx.x, tid = threadIdx.x;
  int b = bj / NN, j = bj - b * NN;
  size_t base = (size_t)b*NNSQ + j;
  float v = (tid < NN) ? la[base + (size_t)tid*NN] : -1e30f;
  float mx = block_reduce256(v, red, tid, 1);
  float e = (tid < NN) ? expf(v - mx) : 0.0f;
  float s = block_reduce256(e, red, tid, 0);
  float lse = mx + logf(s);
  if (tid < NN) la[base + (size_t)tid*NN] = v - lse;
}

__global__ void k_x0(const float* __restrict__ la, float* __restrict__ out){
  int idx = blockIdx.x * 256 + threadIdx.x;
  if (idx >= BSZ * NNSQ) return;
  int b = idx / NNSQ, r = idx - b * NNSQ;
  int i2 = r / NN, j = r - i2 * NN;
  float a = expf(la[idx]);
  float bt = expf(la[(size_t)b*NNSQ + (size_t)j*NN + i2]);
  out[idx] = 0.5f * (a + bt);
}

// ---------------- host ----------------
// Input order: x, timesteps, then params leaves in JAX pytree (alphabetical) order:
// final(b,w), final_mod(b,w), in_proj(b,w),
// layers[0..3]{mlp1(b,w), mlp2(b,w), mod(b,w), wk(b,w), wo(b,w), wq(b,w), wv(b,w)},
// pair[0](b,w), pair[1](b,w), pair_out(b,w), pos_emb, t_mlp1(b,w), t_mlp2(b,w)
enum { LE_MLP1_B=0, LE_MLP1_W, LE_MLP2_B, LE_MLP2_W, LE_MOD_B, LE_MOD_W,
       LE_WK_B, LE_WK_W, LE_WO_B, LE_WO_W, LE_WQ_B, LE_WQ_W, LE_WV_B, LE_WV_W };
#define LPAR(l,e) ((const float*)d_in[8 + 14*(l) + (e)])

extern "C" void kernel_launch(void* const* d_in, const int* in_sizes, int n_in,
                              void* d_out, int out_size, void* d_ws, size_t ws_size,
                              hipStream_t stream)
{
  (void)in_sizes; (void)out_size; (void)ws_size;
  if (n_in < 75) return;

  const float* x_in    = (const float*)d_in[0];
  const int*   ts      = (const int*)  d_in[1];
  const float* final_b = (const float*)d_in[2];
  const float* final_w = (const float*)d_in[3];
  const float* fmod_b  = (const float*)d_in[4];
  const float* fmod_w  = (const float*)d_in[5];
  const float* inp_b   = (const float*)d_in[6];
  const float* inp_w   = (const float*)d_in[7];
  const float* pair0_b = (const float*)d_in[64];
  const float* pair0_w = (const float*)d_in[65];
  const float* pair1_b = (const float*)d_in[66];
  const float* pair1_w = (const float*)d_in[67];
  const float* pout_b  = (const float*)d_in[68];
  const float* pout_w  = (const float*)d_in[69];
  const float* pos_emb = (const float*)d_in[70];
  const float* tm1_b   = (const float*)d_in[71];
  const float* tm1_w   = (const float*)d_in[72];
  const float* tm2_b   = (const float*)d_in[73];
  const float* tm2_w   = (const float*)d_in[74];

  float* ws   = (float*)d_ws;
  float* tbuf = ws;  ws += BSZ*DD;           // 2048
  float* st   = ws;  ws += BSZ*DD;           // silu(t)
  float* mods = ws;  ws += 4*BSZ*6*DD;       // per-layer modulation [4][8][1536]
  float* fmod = ws;  ws += BSZ*2*DD;         // final modulation [8][512]
  float* h    = ws;  ws += NROWS*DD;
  float* hn   = ws;  ws += NROWS*DD;
  float* qb   = ws;  ws += NROWS*DD;
  float* kb   = ws;  ws += NROWS*DD;
  float* vb   = ws;  ws += NROWS*DD;
  float* ao   = ws;  ws += NROWS*DD;
  float* m1   = ws;  ws += NROWS*4*DD;
  float* hp   = ws;  ws += NROWS*DD;
  float* uu   = ws;  ws += NROWS*DD;
  float* vv   = ws;  ws += NROWS*DD;
  float* Lraw = ws;  ws += BSZ*NNSQ;
  float* la   = ws;  ws += BSZ*NNSQ;
  _Float16* w2th = (_Float16*)ws;            // 65536 halfs
  _Float16* w2tl = w2th + 256*256;           // 65536 halfs

  float* logits_out = (float*)d_out;

  // timestep embedding + t-MLP + silu(t)
  k_timestep<<<BSZ, 256, 0, stream>>>(ts, tm1_w, tm1_b, tm2_w, tm2_b, tbuf, st);

  // adaLN modulation vectors for all layers + final
  for (int l = 0; l < 4; ++l)
    k_smallmm<<<(BSZ*6*DD + 255)/256, 256, 0, stream>>>(st, LPAR(l, LE_MOD_W), LPAR(l, LE_MOD_B),
                                                        mods + l*BSZ*6*DD, 6*DD);
  k_smallmm<<<(BSZ*2*DD + 255)/256, 256, 0, stream>>>(st, fmod_w, fmod_b, fmod, 2*DD);

  // h = x @ in_proj + b + pos_emb
  k_gemm<<<dim3(DD/64, NROWS/64), 256, 0, stream>>>(x_in, inp_w, inp_b, h,
      NROWS, ENCD, DD, 1, pos_emb, nullptr);

  for (int l = 0; l < 4; ++l){
    float* mod_l = mods + l*BSZ*6*DD;        // [8][1536]: s1|sc1|g1|s2|sc2|g2
    k_ln_mod<<<NROWS, 256, 0, stream>>>(h, hn, mod_l, mod_l + DD, 6*DD);
    k_gemm<<<dim3(DD/64, NROWS/64), 256, 0, stream>>>(hn, LPAR(l, LE_WQ_W), LPAR(l, LE_WQ_B), qb,
        NROWS, DD, DD, 0, nullptr, nullptr);
    k_gemm<<<dim3(DD/64, NROWS/64), 256, 0, stream>>>(hn, LPAR(l, LE_WK_W), LPAR(l, LE_WK_B), kb,
        NROWS, DD, DD, 0, nullptr, nullptr);
    k_gemm<<<dim3(DD/64, NROWS/64), 256, 0, stream>>>(hn, LPAR(l, LE_WV_W), LPAR(l, LE_WV_B), vb,
        NROWS, DD, DD, 0, nullptr, nullptr);
    k_attn<<<BSZ*4, NN, 2*NN*HD*sizeof(float), stream>>>(qb, kb, vb, ao);
    k_gemm<<<dim3(DD/64, NROWS/64), 256, 0, stream>>>(ao, LPAR(l, LE_WO_W), LPAR(l, LE_WO_B), h,
        NROWS, DD, DD, 2, mod_l + 2*DD, h);
    k_ln_mod<<<NROWS, 256, 0, stream>>>(h, hn, mod_l + 3*DD, mod_l + 4*DD, 6*DD);
    k_gemm<<<dim3(4*DD/64, NROWS/64), 256, 0, stream>>>(hn, LPAR(l, LE_MLP1_W), LPAR(l, LE_MLP1_B), m1,
        NROWS, DD, 4*DD, 3, nullptr, nullptr);
    k_gemm<<<dim3(DD/64, NROWS/64), 256, 0, stream>>>(m1, LPAR(l, LE_MLP2_W), LPAR(l, LE_MLP2_B), h,
        NROWS, 4*DD, DD, 2, mod_l + 5*DD, h);
  }

  // final adaLN + projection
  k_ln_mod<<<NROWS, 256, 0, stream>>>(h, hn, fmod, fmod + DD, 2*DD);
  k_gemm<<<dim3(DD/64, NROWS/64), 256, 0, stream>>>(hn, final_w, final_b, hp,
      NROWS, DD, DD, 0, nullptr, nullptr);

  // pair layer-1 split: u = hp@W_top + b1 ; v = hp@W_bot
  k_gemm<<<dim3(DD/64, NROWS/64), 256, 0, stream>>>(hp, pair0_w, pair0_b, uu,
      NROWS, DD, DD, 0, nullptr, nullptr);
  k_gemm<<<dim3(DD/64, NROWS/64), 256, 0, stream>>>(hp, pair0_w + 256*256, nullptr, vv,
      NROWS, DD, DD, 0, nullptr, nullptr);

  // pair layer-2 weights -> K-major f16 hi/lo once; then fused pair kernel
  k_pack_w2<<<256, 256, 0, stream>>>(pair1_w, w2th, w2tl);
  size_t pair_lds = (size_t)(2*96*256*sizeof(_Float16)) + NN*sizeof(float);
  k_pair<<<NROWS, 256, pair_lds, stream>>>(uu, vv, w2th, w2tl, pair1_b, pout_w, pout_b, Lraw);

  // symmetrize + mask + output logits + Sinkhorn
  k_sym<<<(BSZ*NNSQ + 255)/256, 256, 0, stream>>>(Lraw, logits_out, la);
  for (int it = 0; it < 20; ++it){
    k_sink_row<<<NROWS, 256, 0, stream>>>(la);
    k_sink_col<<<NROWS, 256, 0, stream>>>(la);
  }
  k_x0<<<(BSZ*NNSQ + 255)/256, 256, 0, stream>>>(la, logits_out + BSZ*NNSQ);
}
